// StableGATModel_50955492000376
// MI455X (gfx1250) — matrix operand order repaired and checked by reading, not compiled, not run on hardware
//
#include <hip/hip_runtime.h>
#include <stdint.h>

#define HEADS 2
#define HID   64
#define HF    128   // HEADS*HID
#define OUTF  16
#define BLK   256

typedef __attribute__((ext_vector_type(16))) __bf16 v16bf;
typedef __attribute__((ext_vector_type(8)))  float  v8f;
typedef int v4i_ __attribute__((vector_size(16)));   // matches async-LDS builtin param

#if defined(__HIP_DEVICE_COMPILE__) && defined(__has_builtin)
#if __has_builtin(__builtin_amdgcn_global_load_async_to_lds_b128)
#define HAVE_ASYNC_LDS 1
#endif
#endif

__device__ __forceinline__ uint16_t f2bf(float f) {
  uint32_t u = __float_as_uint(f);
  u += 0x7FFFu + ((u >> 16) & 1u);          // round-to-nearest-even
  return (uint16_t)(u >> 16);
}

__device__ __forceinline__ void atomicAddF(float* p, float v) {
  __hip_atomic_fetch_add(p, v, __ATOMIC_RELAXED, __HIP_MEMORY_SCOPE_AGENT);
}

// ---------------- elementwise / setup kernels ----------------

__global__ void f32_to_bf16_kernel(const float* __restrict__ in,
                                   uint16_t* __restrict__ out, int n) {
  int i = blockIdx.x * blockDim.x + threadIdx.x;
  if (i < n) out[i] = f2bf(in[i]);
}

// Repack W [K,M] f32 row-major into WMMA B-fragment bf16 dwords, chunk-major:
//   fragment linear id g = (chunk*TM + ct)*KCS + ki   (KCS = 4 ksteps per 128-K chunk)
//   dword index        = g*256 + lane*8 + v           (per-lane contiguous -> b128 loads)
// Within a fragment: N=lane%16, K = ks*32 + 16*(v>>2) + 8*(lane>>4) + 2*(v&3), +1 (lo, hi halves).
__global__ void pack_b_kernel(const float* __restrict__ W,
                              uint32_t* __restrict__ Bp, int K, int M) {
  int idx = blockIdx.x * blockDim.x + threadIdx.x;
  const int KCS = 4;
  int ksteps = K >> 5;
  int tm = M >> 4;
  int total = tm * ksteps * 256;
  if (idx >= total) return;
  int g    = idx >> 8;
  int lane = (idx >> 3) & 31;
  int v    = idx & 7;
  int ki    = g % KCS;
  int ct    = (g / KCS) % tm;
  int chunk = g / (KCS * tm);
  int ks = chunk * KCS + ki;
  int n  = ct * 16 + (lane & 15);
  int k0 = ks * 32 + ((v >> 2) << 4) + ((lane >> 4) << 3) + 2 * (v & 3);
  uint32_t lo = f2bf(W[(size_t)k0 * M + n]);
  uint32_t hi = f2bf(W[(size_t)(k0 + 1) * M + n]);
  Bp[idx] = lo | (hi << 16);
}

__global__ void init_nodes_kernel(int* __restrict__ mkey, float* __restrict__ S, int n2) {
  int i = blockIdx.x * blockDim.x + threadIdx.x;
  if (i < n2) { mkey[i] = (int)0x80000000; S[i] = 0.f; }
}

__global__ void zero_f32_kernel(float* __restrict__ p, int n) {
  int i = blockIdx.x * blockDim.x + threadIdx.x;
  if (i < n) p[i] = 0.f;
}

// ---------------- WMMA GEMM: C[N,M] = bf16(A[N,K]) @ bf16(W[K,M]) (+bias) ----------------
// One wave per 16-row tile computing all TM column tiles; B chunk staged in LDS
// (async global->LDS when available) and shared by the block's 8 waves.

template<int TM>
__global__ __launch_bounds__(256) void gemm_wmma_bf16(
    const uint16_t* __restrict__ A, const uint32_t* __restrict__ Bp,
    float* __restrict__ C, const float* __restrict__ bias, int N, int K) {
  constexpr int M   = TM * 16;
  constexpr int KCS = 4;                      // k-steps per chunk (128 K values)
  constexpr int CHUNK_DW = TM * KCS * 256;    // dwords of B per chunk
  __shared__ uint32_t sB[CHUNK_DW];

  int tid  = threadIdx.x;
  int lane = tid & 31;
  int nRowTiles = N >> 4;
  int rowTile = blockIdx.x * 8 + (tid >> 5);
  bool active = rowTile < nRowTiles;
  if (!active) rowTile = nRowTiles - 1;       // keep wave alive for barriers/WMMA

  int half = lane >> 4;
  const uint32_t* __restrict__ rowp =
      (const uint32_t*)(A + (size_t)(rowTile * 16 + (lane & 15)) * K);

  v8f acc[TM];
#pragma unroll
  for (int t = 0; t < TM; ++t) acc[t] = (v8f){0.f,0.f,0.f,0.f,0.f,0.f,0.f,0.f};

  int nChunks = K >> 7;                       // K multiple of 128
  for (int chunk = 0; chunk < nChunks; ++chunk) {
    const uint32_t* __restrict__ gsrc = Bp + (size_t)chunk * CHUNK_DW;
    // ---- stage B chunk into LDS (16B per thread per iter, coalesced) ----
#pragma unroll
    for (int j = 0; j < TM; ++j) {
      int idx4 = tid + j * 256;               // float4 index within chunk
#if defined(HAVE_ASYNC_LDS)
      __builtin_amdgcn_global_load_async_to_lds_b128(
          (__attribute__((address_space(1))) v4i_*)(gsrc + idx4 * 4),
          (__attribute__((address_space(3))) v4i_*)(&sB[idx4 * 4]), 0, 0);
#else
      *(float4*)(&sB[idx4 * 4]) = *(const float4*)(gsrc + idx4 * 4);
#endif
    }
#if defined(HAVE_ASYNC_LDS)
#if __has_builtin(__builtin_amdgcn_s_wait_asynccnt)
    __builtin_amdgcn_s_wait_asynccnt(0);
#else
    asm volatile("s_wait_asynccnt 0" ::: "memory");
#endif
#endif
    __syncthreads();

    // ---- compute: KCS k-steps x TM column tiles ----
#pragma unroll
    for (int ki = 0; ki < KCS; ++ki) {
      int kb = (chunk * KCS + ki) << 5;
      union { uint32_t u[8]; v16bf v; } af;
      int b0 = (kb + half * 8) >> 1;
      int b1 = (kb + 16 + half * 8) >> 1;
#pragma unroll
      for (int v = 0; v < 4; ++v) af.u[v]     = rowp[b0 + v];
#pragma unroll
      for (int v = 0; v < 4; ++v) af.u[4 + v] = rowp[b1 + v];
#pragma unroll
      for (int ct = 0; ct < TM; ++ct) {
        union { uint32_t u[8]; v16bf v; } bfr;
        const uint32_t* ls = &sB[(ct * KCS + ki) * 256 + lane * 8];
#pragma unroll
        for (int v = 0; v < 8; ++v) bfr.u[v] = ls[v];
        acc[ct] = __builtin_amdgcn_wmma_f32_16x16x32_bf16(
            false, af.v, false, bfr.v, (short)0, acc[ct], false, false);
      }
    }
    __syncthreads();
  }

  if (active) {
    int rbase = rowTile * 16 + half * 8;
#pragma unroll
    for (int ct = 0; ct < TM; ++ct) {
      int ncol = ct * 16 + (lane & 15);
      float badd = bias ? bias[ncol] : 0.f;
#pragma unroll
      for (int r = 0; r < 8; ++r)
        C[(size_t)(rbase + r) * M + ncol] = acc[ct][r] + badd;
    }
  }
}

// ---------------- GAT attention kernels ----------------

__global__ void scores_kernel(const float* __restrict__ H,
                              const float* __restrict__ al,
                              const float* __restrict__ ar,
                              float* __restrict__ el, float* __restrict__ er, int N) {
  int i = blockIdx.x * blockDim.x + threadIdx.x;
  if (i >= N * HEADS) return;
  int n = i >> 1, h = i & 1;
  const float* hp  = H  + (size_t)n * HF + h * HID;
  const float* alp = al + h * HID;
  const float* arp = ar + h * HID;
  float sl = 0.f, sr = 0.f;
#pragma unroll 8
  for (int f = 0; f < HID; ++f) { float x = hp[f]; sl += x * alp[f]; sr += x * arp[f]; }
  el[i] = sl; er[i] = sr;
}

__device__ __forceinline__ float leaky02(float v) { return v > 0.f ? v : 0.2f * v; }

__global__ void edge_max_kernel(const int* __restrict__ src, const int* __restrict__ dst,
                                const float* __restrict__ el, const float* __restrict__ er,
                                int* __restrict__ mkey, int E) {
  int i = blockIdx.x * blockDim.x + threadIdx.x;
  if (i >= E * HEADS) return;
  int e = i >> 1, h = i & 1;
  int dN = dst[e];
  float v = leaky02(el[src[e] * HEADS + h] + er[dN * HEADS + h]);
  int bits = __float_as_int(v);
  int key  = bits >= 0 ? bits : (bits ^ 0x7FFFFFFF);   // order-preserving int map
  atomicMax(&mkey[dN * HEADS + h], key);
}

__global__ void finalize_m_kernel(const int* __restrict__ mkey,
                                  float* __restrict__ Mv, int n2) {
  int i = blockIdx.x * blockDim.x + threadIdx.x;
  if (i >= n2) return;
  int k = mkey[i];
  float v = 0.f;
  if (k != (int)0x80000000) {
    int bits = k >= 0 ? k : (k ^ 0x7FFFFFFF);
    v = __int_as_float(bits);
    if (!isfinite(v)) v = 0.f;
  }
  Mv[i] = v;
}

__global__ void edge_exp_kernel(const int* __restrict__ src, const int* __restrict__ dst,
                                const float* __restrict__ el, const float* __restrict__ er,
                                const float* __restrict__ Mv, float* __restrict__ S,
                                float* __restrict__ P, int E) {
  int i = blockIdx.x * blockDim.x + threadIdx.x;
  if (i >= E * HEADS) return;
  int e = i >> 1, h = i & 1;
  int dN = dst[e];
  float v = leaky02(el[src[e] * HEADS + h] + er[dN * HEADS + h]);
  float p = __expf(v - Mv[dN * HEADS + h]);
  P[i] = p;
  atomicAddF(&S[dN * HEADS + h], p);
}

// one thread per (edge, 4-feature chunk): E*32 threads
__global__ void edge_agg_kernel(const int* __restrict__ src, const int* __restrict__ dst,
                                const float* __restrict__ P, const float* __restrict__ S,
                                const float* __restrict__ H, float* __restrict__ AGG, int E) {
  long long t = (long long)blockIdx.x * blockDim.x + threadIdx.x;
  if (t >= (long long)E * 32) return;
  int e = (int)(t >> 5);
  int c = (int)(t & 31);
  int h = c >> 4;                       // chunks 0-15 -> head0, 16-31 -> head1
  int sN = src[e], dN = dst[e];
  float alpha = P[e * HEADS + h] / (S[dN * HEADS + h] + 1e-9f);
  const float4 hv = *(const float4*)(H + (size_t)sN * HF + c * 4);
  float* op = AGG + (size_t)dN * HF + c * 4;
  atomicAddF(op + 0, alpha * hv.x);
  atomicAddF(op + 1, alpha * hv.y);
  atomicAddF(op + 2, alpha * hv.z);
  atomicAddF(op + 3, alpha * hv.w);
}

__global__ void epilogue_kernel(const float* __restrict__ AGG,
                                const float* __restrict__ b,
                                uint16_t* __restrict__ Abf, int N) {
  int i = blockIdx.x * blockDim.x + threadIdx.x;
  if (i >= N * HF) return;
  float v = AGG[i] + b[i % HF];
  v = v > 0.f ? v : expm1f(v);          // ELU(alpha=1)
  Abf[i] = f2bf(v);
}

// ---------------- launch ----------------

static inline int nblk(long long n) { return (int)((n + BLK - 1) / BLK); }

extern "C" void kernel_launch(void* const* d_in, const int* in_sizes, int n_in,
                              void* d_out, int out_size, void* d_ws, size_t ws_size,
                              hipStream_t stream) {
  (void)n_in; (void)out_size; (void)ws_size;
  const float* features = (const float*)d_in[0];
  const int*   src      = (const int*)d_in[1];
  const int*   dst      = (const int*)d_in[2];
  const float* W1 = (const float*)d_in[3];
  const float* al1 = (const float*)d_in[4];
  const float* ar1 = (const float*)d_in[5];
  const float* b1  = (const float*)d_in[6];
  const float* W2 = (const float*)d_in[7];
  const float* al2 = (const float*)d_in[8];
  const float* ar2 = (const float*)d_in[9];
  const float* b2  = (const float*)d_in[10];
  const float* Wl = (const float*)d_in[11];
  const float* bl = (const float*)d_in[12];

  const int IN = 256;
  const int N  = in_sizes[0] / IN;      // 100000
  const int E  = in_sizes[1];           // 1600000

  size_t off = 0;
  auto take = [&](size_t bytes) -> void* {
    void* p = (char*)d_ws + off;
    off += (bytes + 255) & ~(size_t)255;
    return p;
  };
  uint16_t* A16  = (uint16_t*)take((size_t)N * IN * 2);   // current GEMM input (bf16)
  float*    H    = (float*)take((size_t)N * HF * 4);      // GEMM output h
  float*    AGG  = (float*)take((size_t)N * HF * 4);      // aggregation accumulator
  float*    el   = (float*)take((size_t)N * HEADS * 4);
  float*    er   = (float*)take((size_t)N * HEADS * 4);
  int*      mkey = (int*)take((size_t)N * HEADS * 4);
  float*    Mv   = (float*)take((size_t)N * HEADS * 4);
  float*    S    = (float*)take((size_t)N * HEADS * 4);
  float*    P    = (float*)take((size_t)E * HEADS * 4);
  uint32_t* W1p  = (uint32_t*)take((size_t)(HF/16)*(IN/32)*256*4);
  uint32_t* W2p  = (uint32_t*)take((size_t)(HF/16)*(HF/32)*256*4);
  uint32_t* Wlp  = (uint32_t*)take((size_t)(OUTF/16)*(HF/32)*256*4);

  // weight repacks (deterministic, redone every call)
  pack_b_kernel<<<nblk((HF/16)*(IN/32)*256), BLK, 0, stream>>>(W1, W1p, IN, HF);
  pack_b_kernel<<<nblk((HF/16)*(HF/32)*256), BLK, 0, stream>>>(W2, W2p, HF, HF);
  pack_b_kernel<<<nblk((OUTF/16)*(HF/32)*256), BLK, 0, stream>>>(Wl, Wlp, HF, OUTF);
  f32_to_bf16_kernel<<<nblk((long long)N * IN), BLK, 0, stream>>>(features, A16, N * IN);

  const float* als[2] = {al1, al2};
  const float* ars[2] = {ar1, ar2};
  const float* bs[2]  = {b1, b2};
  const uint32_t* Wps[2] = {W1p, W2p};
  const int Ks[2] = {IN, HF};

  int gemmBlocks = (N / 16 + 7) / 8;    // 8 row-tile waves per block

  for (int layer = 0; layer < 2; ++layer) {
    gemm_wmma_bf16<HF/16><<<gemmBlocks, BLK, 0, stream>>>(
        A16, Wps[layer], H, nullptr, N, Ks[layer]);
    scores_kernel<<<nblk((long long)N * HEADS), BLK, 0, stream>>>(
        H, als[layer], ars[layer], el, er, N);
    init_nodes_kernel<<<nblk((long long)N * HEADS), BLK, 0, stream>>>(mkey, S, N * HEADS);
    zero_f32_kernel<<<nblk((long long)N * HF), BLK, 0, stream>>>(AGG, N * HF);
    edge_max_kernel<<<nblk((long long)E * HEADS), BLK, 0, stream>>>(src, dst, el, er, mkey, E);
    finalize_m_kernel<<<nblk((long long)N * HEADS), BLK, 0, stream>>>(mkey, Mv, N * HEADS);
    edge_exp_kernel<<<nblk((long long)E * HEADS), BLK, 0, stream>>>(src, dst, el, er, Mv, S, P, E);
    edge_agg_kernel<<<nblk((long long)E * 32), BLK, 0, stream>>>(src, dst, P, S, H, AGG, E);
    epilogue_kernel<<<nblk((long long)N * HF), BLK, 0, stream>>>(AGG, bs[layer], A16, N);
  }

  // final linear head: d_out[N,16] = act @ Wl + bl
  gemm_wmma_bf16<OUTF/16><<<gemmBlocks, BLK, 0, stream>>>(
      A16, Wlp, (float*)d_out, bl, N, HF);
}
